// GraphEmbeddings_72576357367893
// MI455X (gfx1250) — matrix operand order repaired and checked by reading, not compiled, not run on hardware
//
#include <hip/hip_runtime.h>
#include <stdint.h>

// ---------------- problem constants ----------------
constexpr int NN  = 32768;    // nodes
constexpr int EE  = 262144;   // edges (without self loops)
constexpr int EPn = EE + NN;  // edges + self loops = 294912
constexpr int TT  = 32;       // nodes per graph
constexpr int BB  = NN / TT;  // graphs = 1024
constexpr float NEG_SLOPE = 0.2f;
constexpr float BN_EPS    = 1e-5f;

// ---------------- workspace layout (bytes) ----------------
constexpr size_t OFF_DEG    = 0;                               // N f32
constexpr size_t OFF_EALOOP = OFF_DEG    + (size_t)NN * 4;     // N*4 f32
constexpr size_t OFF_MAXK   = OFF_EALOOP + (size_t)NN * 16;    // N*2 u32
constexpr size_t OFF_DENOM  = OFF_MAXK   + (size_t)NN * 8;     // N*2 f32
constexpr size_t OFF_LOGIT  = OFF_DENOM  + (size_t)NN * 8;     // EP*2 f32
constexpr size_t OFF_BNST   = OFF_LOGIT  + (size_t)EPn * 8;    // 512 f32
constexpr size_t OFF_PWL2   = OFF_BNST   + 4096;               // 256x256 bf16
constexpr size_t OFF_PWR2   = OFF_PWL2   + 131072;
constexpr size_t OFF_PWL3   = OFF_PWR2   + 131072;             // 128x256 bf16
constexpr size_t OFF_PWR3   = OFF_PWL3   + 65536;
constexpr size_t OFF_PWIH   = OFF_PWR3   + 65536;              // 128x384 bf16
constexpr size_t OFF_PWHH   = OFF_PWIH   + 98304;
constexpr size_t OFF_XL     = (size_t)8 << 20;                 // N*256 f32 (also HM)
constexpr size_t OFF_AGG    = OFF_XL  + (size_t)NN * 256 * 4;  // N*256 f32 (aliases XR)
constexpr size_t OFF_ABF    = OFF_AGG + (size_t)NN * 256 * 4;  // N*256 bf16

// ---------------- WMMA types / helpers ----------------
typedef __attribute__((ext_vector_type(16))) __bf16 v16bf;
typedef __attribute__((ext_vector_type(8)))  float  v8f;

struct B32x8 { uint4 lo, hi; };

__device__ __forceinline__ v16bf load_frag16(const uint16_t* p0, const uint16_t* p1) {
    B32x8 r;
    r.lo = *(const uint4*)p0;   // 8 bf16
    r.hi = *(const uint4*)p1;   // 8 bf16
    return __builtin_bit_cast(v16bf, r);
}

__device__ __forceinline__ uint16_t f2bf(float f) {
    unsigned u = __float_as_uint(f);
    unsigned r = u + 0x7FFFu + ((u >> 16) & 1u);   // round-to-nearest-even
    return (uint16_t)(r >> 16);
}

// order-preserving float->uint key for atomicMax-based segment max
__device__ __forceinline__ unsigned fkey(float f) {
    unsigned u = __float_as_uint(f);
    return (u & 0x80000000u) ? ~u : (u ^ 0x80000000u);
}
__device__ __forceinline__ float funkey(unsigned k) {
    unsigned u = (k & 0x80000000u) ? (k ^ 0x80000000u) : ~k;
    return __uint_as_float(u);
}

// ---------------- self-loop prep ----------------
__global__ void deg_easum_kernel(const int* __restrict__ eidx,
                                 const float* __restrict__ edge_attr,
                                 float* __restrict__ deg, float* __restrict__ easum) {
    int e = blockIdx.x * blockDim.x + threadIdx.x;
    if (e >= EE) return;
    int d = eidx[EE + e];
    atomicAdd(&deg[d], 1.0f);
    atomicAdd(&easum[(size_t)d * 4 + 0], edge_attr[(size_t)e * 4 + 0]);
    atomicAdd(&easum[(size_t)d * 4 + 1], edge_attr[(size_t)e * 4 + 1]);
    atomicAdd(&easum[(size_t)d * 4 + 2], edge_attr[(size_t)e * 4 + 2]);
    atomicAdd(&easum[(size_t)d * 4 + 3], edge_attr[(size_t)e * 4 + 3]);
}

__global__ void ea_mean_kernel(const float* __restrict__ deg, float* __restrict__ ea) {
    int i = blockIdx.x * blockDim.x + threadIdx.x;
    if (i >= NN * 4) return;
    ea[i] /= fmaxf(deg[i >> 2], 1.0f);
}

// ---------------- weight pre-pack into WMMA B-fragment order ----------------
// frag f = kt*NT + nt; element idx = ((f*32 + lane)*16 + j)
// B[k][n], k = kt*32 + (lane<16?0:8) + (j<8 ? j : 16 + j - 8), n = nt*16 + (lane&15)
__global__ void pack_b_kernel(const float* __restrict__ W, uint16_t* __restrict__ out,
                              int K, int Nc, int transpose) {
    int idx = blockIdx.x * blockDim.x + threadIdx.x;
    if (idx >= K * Nc) return;
    int NT = Nc >> 4;
    int f = idx >> 9, r = idx & 511;
    int lane = r >> 4, j = r & 15;
    int kt = f / NT, nt = f % NT;
    int k = kt * 32 + ((lane < 16) ? 0 : 8) + ((j < 8) ? j : 16 + (j - 8));
    int n = nt * 16 + (lane & 15);
    float v = transpose ? W[(size_t)n * K + k] : W[(size_t)k * Nc + n];
    out[idx] = f2bf(v);
}

// ---------------- bf16 WMMA GEMM: C[M,Nc] = A[M,K] * B ----------------
__global__ __launch_bounds__(256) void gemm_bf16_wmma(const uint16_t* __restrict__ A,
                                                      const uint16_t* __restrict__ Bp,
                                                      float* __restrict__ C,
                                                      int M, int K, int Nc) {
    const int lane = threadIdx.x & 31;
    const int wave = blockIdx.x * (blockDim.x >> 5) + (threadIdx.x >> 5);
    const int NT = Nc >> 4;
    const int mtile = wave / NT, ntile = wave % NT;
    if (mtile * 16 >= M) return;
    const int m  = mtile * 16 + (lane & 15);
    const int kh = (lane < 16) ? 0 : 8;
    v8f acc = {};
    const int KT = K >> 5;
    for (int kt = 0; kt < KT; ++kt) {
        const uint16_t* a0 = A + (size_t)m * K + kt * 32 + kh;
        __builtin_prefetch(a0 + 32, 0, 1);
        v16bf af = load_frag16(a0, a0 + 16);
        const uint16_t* b0 = Bp + ((size_t)(kt * NT + ntile) * 32 + lane) * 16;
        v16bf bfv = load_frag16(b0, b0 + 8);
        acc = __builtin_amdgcn_wmma_f32_16x16x32_bf16(false, af, false, bfv,
                                                      (short)0, acc, false, false);
    }
    const int row0 = mtile * 16 + ((lane < 16) ? 0 : 8);
    const int col  = ntile * 16 + (lane & 15);
#pragma unroll
    for (int r = 0; r < 8; ++r) C[(size_t)(row0 + r) * Nc + col] = acc[r];
}

// ---------------- layer 1 node transforms (rank-1, in=1) ----------------
__global__ void layer1_xlxr(const float* __restrict__ x, const float* __restrict__ Wl,
                            const float* __restrict__ Wr,
                            float* __restrict__ XL, float* __restrict__ XR) {
    size_t i = (size_t)blockIdx.x * blockDim.x + threadIdx.x;
    if (i >= (size_t)NN * 256) return;
    int n = (int)(i >> 8), c = (int)(i & 255);
    float xv = x[n];
    XL[i] = xv * Wl[c];
    XR[i] = xv * Wr[c];
}

// ---------------- edge logits: wave per edge ----------------
__global__ __launch_bounds__(256) void edge_logit_kernel(
        const float* __restrict__ XL, const float* __restrict__ XR,
        const int* __restrict__ eidx, const float* __restrict__ edge_attr,
        const float* __restrict__ ealoop, const float* __restrict__ We,
        const float* __restrict__ att, float* __restrict__ logit, int hshift) {
    const int lane = threadIdx.x & 31;
    const int e = blockIdx.x * (blockDim.x >> 5) + (threadIdx.x >> 5);
    if (e >= EPn) return;
    int s, d; const float* ea;
    if (e < EE) { s = eidx[e]; d = eidx[EE + e]; ea = edge_attr + (size_t)e * 4; }
    else        { s = d = e - EE;                ea = ealoop   + (size_t)s * 4; }
    float ea0 = ea[0], ea1 = ea[1], ea2 = ea[2], ea3 = ea[3];
    const float* xl = XL + (size_t)s * 256;
    const float* xr = XR + (size_t)d * 256;
    float acc0 = 0.f, acc1 = 0.f;
#pragma unroll
    for (int j = 0; j < 8; ++j) {
        int c = lane + 32 * j;
        float el = ea0 * We[c] + ea1 * We[256 + c] + ea2 * We[512 + c] + ea3 * We[768 + c];
        float v = xl[c] + xr[c] + el;
        v = (v > 0.f) ? v : v * NEG_SLOPE;
        float contrib = v * att[c];
        if ((c >> hshift) == 0) acc0 += contrib; else acc1 += contrib;
    }
    for (int off = 16; off; off >>= 1) {
        acc0 += __shfl_down(acc0, off, 32);
        acc1 += __shfl_down(acc1, off, 32);
    }
    if (lane == 0) {
        int H = (hshift == 7) ? 2 : 1;
        logit[(size_t)e * H] = acc0;
        if (H == 2) logit[(size_t)e * H + 1] = acc1;
    }
}

// ---------------- segment softmax ----------------
__global__ void seg_max_kernel(const float* __restrict__ logit, const int* __restrict__ eidx,
                               unsigned* __restrict__ maxk, int H) {
    int i = blockIdx.x * blockDim.x + threadIdx.x;
    if (i >= EPn * H) return;
    int e = i / H, h = i % H;
    int d = (e < EE) ? eidx[EE + e] : (e - EE);
    atomicMax(&maxk[d * H + h], fkey(logit[i]));
}

__global__ void expsum_kernel(float* __restrict__ logit, const int* __restrict__ eidx,
                              const unsigned* __restrict__ maxk, float* __restrict__ denom, int H) {
    int i = blockIdx.x * blockDim.x + threadIdx.x;
    if (i >= EPn * H) return;
    int e = i / H, h = i % H;
    int d = (e < EE) ? eidx[EE + e] : (e - EE);
    float p = __expf(logit[i] - funkey(maxk[d * H + h]));
    logit[i] = p;
    atomicAdd(&denom[d * H + h], p);
}

// ---------------- aggregation scatter: wave per edge ----------------
__global__ __launch_bounds__(256) void aggregate_kernel(
        const float* __restrict__ XL, const float* __restrict__ p,
        const float* __restrict__ denom, const int* __restrict__ eidx,
        float* __restrict__ AGG, int H, int hshift) {
    const int lane = threadIdx.x & 31;
    const int e = blockIdx.x * (blockDim.x >> 5) + (threadIdx.x >> 5);
    if (e >= EPn) return;
    int s, d;
    if (e < EE) { s = eidx[e]; d = eidx[EE + e]; } else { s = d = e - EE; }
    float a0 = p[(size_t)e * H] / denom[d * H];
    float a1 = (H == 2) ? p[(size_t)e * H + 1] / denom[d * H + 1] : 0.f;
    const float* xl = XL + (size_t)s * 256;
    float* out = AGG + (size_t)d * 256;
#pragma unroll
    for (int j = 0; j < 8; ++j) {
        int c = lane + 32 * j;
        float a = (c >> hshift) ? a1 : a0;
        atomicAdd(&out[c], xl[c] * a);
    }
}

// ---------------- head mean + bias ----------------
__global__ void head_mean_bias_kernel(const float* __restrict__ AGG, const float* __restrict__ bias,
                                      float* __restrict__ HM, int outdim) {
    size_t i = (size_t)blockIdx.x * blockDim.x + threadIdx.x;
    if (i >= (size_t)NN * outdim) return;
    int n = (int)(i / outdim), c = (int)(i % outdim);
    float v = (outdim == 256)
                  ? AGG[i]
                  : 0.5f * (AGG[(size_t)n * 256 + c] + AGG[(size_t)n * 256 + 128 + c]);
    HM[i] = v + bias[c];
}

// ---------------- batchnorm (training stats) + relu + f32->bf16 ----------------
__global__ __launch_bounds__(256) void bn_stats_kernel(const float* __restrict__ HM,
                                                       float* __restrict__ stats, int outdim) {
    __shared__ float ss[256], ss2[256];
    int c = blockIdx.x;
    float s = 0.f, s2 = 0.f;
    for (int n = threadIdx.x; n < NN; n += blockDim.x) {
        float v = HM[(size_t)n * outdim + c];
        s += v; s2 += v * v;
    }
    ss[threadIdx.x] = s; ss2[threadIdx.x] = s2;
    __syncthreads();
    for (int o = 128; o; o >>= 1) {
        if ((int)threadIdx.x < o) { ss[threadIdx.x] += ss[threadIdx.x + o];
                                    ss2[threadIdx.x] += ss2[threadIdx.x + o]; }
        __syncthreads();
    }
    if (threadIdx.x == 0) {
        float mu  = ss[0] / (float)NN;
        float var = ss2[0] / (float)NN - mu * mu;
        stats[c * 2 + 0] = mu;
        stats[c * 2 + 1] = rsqrtf(var + BN_EPS);
    }
}

__global__ void bn_apply_kernel(const float* __restrict__ HM, const float* __restrict__ stats,
                                const float* __restrict__ g, const float* __restrict__ be,
                                uint16_t* __restrict__ Abf, int outdim) {
    size_t i = (size_t)blockIdx.x * blockDim.x + threadIdx.x;
    if (i >= (size_t)NN * outdim) return;
    int c = (int)(i % outdim);
    float y = g[c] * (HM[i] - stats[c * 2]) * stats[c * 2 + 1] + be[c];
    Abf[i] = f2bf(fmaxf(y, 0.f));
}

// ---------------- persistent GRU: 64 blocks x 16 graphs, h lives in LDS ----------------
__global__ __launch_bounds__(256) void gru_kernel(const uint16_t* __restrict__ Xbf, // [N,128] bf16
                                                  const uint16_t* __restrict__ pWih,
                                                  const uint16_t* __restrict__ pWhh,
                                                  const float* __restrict__ bih,
                                                  const float* __restrict__ bhh,
                                                  float* __restrict__ out) {
    __shared__ __align__(16) float    GI[16 * 384];
    __shared__ __align__(16) float    GH[16 * 384];
    __shared__ __align__(16) float    hF[16 * 128];
    __shared__ __align__(16) uint16_t hB[16 * 128];
    const int tid = threadIdx.x, lane = tid & 31, wv = tid >> 5;  // 8 waves
    const int g0 = blockIdx.x * 16;
    for (int i = tid; i < 16 * 128; i += 256) { hF[i] = 0.f; hB[i] = 0; }
    __syncthreads();
    const int NT = 24;  // 384/16
    for (int t = 0; t < TT; ++t) {
        // GI (jobs 0..23) and GH (jobs 24..47), 6 jobs per wave
        for (int job = wv * 6; job < wv * 6 + 6; ++job) {
            const bool isGI = job < 24;
            const int ntile = isGI ? job : job - 24;
            const int kh = (lane < 16) ? 0 : 8;
            v8f acc = {};
#pragma unroll
            for (int kt = 0; kt < 4; ++kt) {
                v16bf af;
                if (isGI) {
                    const uint16_t* a0 = Xbf +
                        ((size_t)(g0 + (lane & 15)) * TT + t) * 128 + kt * 32 + kh;
                    af = load_frag16(a0, a0 + 16);
                } else {
                    const uint16_t* a0 = hB + (lane & 15) * 128 + kt * 32 + kh;
                    af = load_frag16(a0, a0 + 16);
                }
                const uint16_t* Bp = isGI ? pWih : pWhh;
                const uint16_t* b0 = Bp + ((size_t)(kt * NT + ntile) * 32 + lane) * 16;
                v16bf bfv = load_frag16(b0, b0 + 8);
                acc = __builtin_amdgcn_wmma_f32_16x16x32_bf16(false, af, false, bfv,
                                                              (short)0, acc, false, false);
            }
            float* O = isGI ? GI : GH;
            const int row0 = (lane < 16) ? 0 : 8;
            const int col  = ntile * 16 + (lane & 15);
#pragma unroll
            for (int r = 0; r < 8; ++r) O[(row0 + r) * 384 + col] = acc[r];
        }
        __syncthreads();
        // gates (torch order r,z,n)
        for (int i = tid; i < 16 * 128; i += 256) {
            int q = i >> 7, c = i & 127;
            float gr  = GI[q * 384 + c]       + bih[c]       + GH[q * 384 + c]       + bhh[c];
            float gz  = GI[q * 384 + 128 + c] + bih[128 + c] + GH[q * 384 + 128 + c] + bhh[128 + c];
            float in_ = GI[q * 384 + 256 + c] + bih[256 + c];
            float hn  = GH[q * 384 + 256 + c] + bhh[256 + c];
            float r = 1.f / (1.f + __expf(-gr));
            float z = 1.f / (1.f + __expf(-gz));
            float nn = tanhf(in_ + r * hn);
            float h = (1.f - z) * nn + z * hF[i];
            hF[i] = h;
            hB[i] = f2bf(h);
        }
        __syncthreads();
    }
    for (int i = tid; i < 16 * 128; i += 256)
        out[(size_t)(g0 + (i >> 7)) * 128 + (i & 127)] = hF[i];
}

// ---------------- host orchestration ----------------
extern "C" void kernel_launch(void* const* d_in, const int* in_sizes, int n_in,
                              void* d_out, int out_size, void* d_ws, size_t ws_size,
                              hipStream_t stream) {
    (void)in_sizes; (void)n_in; (void)out_size; (void)ws_size;
    const float* x         = (const float*)d_in[0];
    const float* edge_attr = (const float*)d_in[1];
    const int*   eidx      = (const int*)d_in[2];
    // d_in[3] = batch (derivable: node/32), unused
    const float *Wl1=(const float*)d_in[4],  *Wr1=(const float*)d_in[5],  *We1=(const float*)d_in[6];
    const float *att1=(const float*)d_in[7], *b1=(const float*)d_in[8],   *g1=(const float*)d_in[9],  *be1=(const float*)d_in[10];
    const float *Wl2=(const float*)d_in[11], *Wr2=(const float*)d_in[12], *We2=(const float*)d_in[13];
    const float *att2=(const float*)d_in[14],*b2=(const float*)d_in[15],  *g2=(const float*)d_in[16], *be2=(const float*)d_in[17];
    const float *Wl3=(const float*)d_in[18], *Wr3=(const float*)d_in[19], *We3=(const float*)d_in[20];
    const float *att3=(const float*)d_in[21],*b3=(const float*)d_in[22],  *g3=(const float*)d_in[23], *be3=(const float*)d_in[24];
    const float *Wih=(const float*)d_in[25], *Whh=(const float*)d_in[26];
    const float *bih=(const float*)d_in[27], *bhh=(const float*)d_in[28];

    char* ws = (char*)d_ws;
    float*    deg    = (float*)(ws + OFF_DEG);
    float*    ealoop = (float*)(ws + OFF_EALOOP);
    unsigned* maxk   = (unsigned*)(ws + OFF_MAXK);
    float*    denom  = (float*)(ws + OFF_DENOM);
    float*    logit  = (float*)(ws + OFF_LOGIT);
    float*    bnst   = (float*)(ws + OFF_BNST);
    uint16_t* pWl2 = (uint16_t*)(ws + OFF_PWL2);
    uint16_t* pWr2 = (uint16_t*)(ws + OFF_PWR2);
    uint16_t* pWl3 = (uint16_t*)(ws + OFF_PWL3);
    uint16_t* pWr3 = (uint16_t*)(ws + OFF_PWR3);
    uint16_t* pWih = (uint16_t*)(ws + OFF_PWIH);
    uint16_t* pWhh = (uint16_t*)(ws + OFF_PWHH);
    float*    XL  = (float*)(ws + OFF_XL);
    float*    AGG = (float*)(ws + OFF_AGG);
    float*    XR  = AGG;  // intentional alias: AGG memset ordered after edge_logit
    uint16_t* Abf = (uint16_t*)(ws + OFF_ABF);
    float*    outp = (float*)d_out;

    // prologue: self-loop attrs + weight packs
    hipMemsetAsync(ws + OFF_DEG, 0, (size_t)NN * 4 + (size_t)NN * 16, stream);
    deg_easum_kernel<<<EE / 256, 256, 0, stream>>>(eidx, edge_attr, deg, ealoop);
    ea_mean_kernel<<<(NN * 4) / 256, 256, 0, stream>>>(deg, ealoop);
    pack_b_kernel<<<(256 * 256) / 256, 256, 0, stream>>>(Wl2, pWl2, 256, 256, 0);
    pack_b_kernel<<<(256 * 256) / 256, 256, 0, stream>>>(Wr2, pWr2, 256, 256, 0);
    pack_b_kernel<<<(128 * 256) / 256, 256, 0, stream>>>(Wl3, pWl3, 128, 256, 0);
    pack_b_kernel<<<(128 * 256) / 256, 256, 0, stream>>>(Wr3, pWr3, 128, 256, 0);
    pack_b_kernel<<<(128 * 384) / 256, 256, 0, stream>>>(Wih, pWih, 128, 384, 1);  // B = Wih^T
    pack_b_kernel<<<(128 * 384) / 256, 256, 0, stream>>>(Whh, pWhh, 128, 384, 1);  // B = Whh^T

    const int EB = EPn / 8;            // wave-per-edge blocks
    const int NC256 = (NN * 256) / 256, NC128 = (NN * 128) / 256;

    // ---------------- layer 1 (in=1, H=1, C=256) ----------------
    layer1_xlxr<<<NC256, 256, 0, stream>>>(x, Wl1, Wr1, XL, XR);
    hipMemsetAsync(maxk, 0, (size_t)NN * 16, stream);  // maxk + denom (contiguous)
    edge_logit_kernel<<<EB, 256, 0, stream>>>(XL, XR, eidx, edge_attr, ealoop, We1, att1, logit, 8);
    hipMemsetAsync(AGG, 0, (size_t)NN * 256 * 4, stream);
    seg_max_kernel<<<(EPn + 255) / 256, 256, 0, stream>>>(logit, eidx, maxk, 1);
    expsum_kernel<<<(EPn + 255) / 256, 256, 0, stream>>>(logit, eidx, maxk, denom, 1);
    aggregate_kernel<<<EB, 256, 0, stream>>>(XL, logit, denom, eidx, AGG, 1, 8);
    head_mean_bias_kernel<<<NC256, 256, 0, stream>>>(AGG, b1, XL, 256);
    bn_stats_kernel<<<256, 256, 0, stream>>>(XL, bnst, 256);
    bn_apply_kernel<<<NC256, 256, 0, stream>>>(XL, bnst, g1, be1, Abf, 256);

    // ---------------- layer 2 (in=256, H=2, C=128) ----------------
    gemm_bf16_wmma<<<4096, 256, 0, stream>>>(Abf, pWl2, XL, NN, 256, 256);
    gemm_bf16_wmma<<<4096, 256, 0, stream>>>(Abf, pWr2, XR, NN, 256, 256);
    hipMemsetAsync(maxk, 0, (size_t)NN * 16, stream);
    edge_logit_kernel<<<EB, 256, 0, stream>>>(XL, XR, eidx, edge_attr, ealoop, We2, att2, logit, 7);
    hipMemsetAsync(AGG, 0, (size_t)NN * 256 * 4, stream);
    seg_max_kernel<<<(EPn * 2) / 256, 256, 0, stream>>>(logit, eidx, maxk, 2);
    expsum_kernel<<<(EPn * 2) / 256, 256, 0, stream>>>(logit, eidx, maxk, denom, 2);
    aggregate_kernel<<<EB, 256, 0, stream>>>(XL, logit, denom, eidx, AGG, 2, 7);
    head_mean_bias_kernel<<<NC128, 256, 0, stream>>>(AGG, b2, XL, 128);
    bn_stats_kernel<<<128, 256, 0, stream>>>(XL, bnst, 128);
    bn_apply_kernel<<<NC128, 256, 0, stream>>>(XL, bnst, g2, be2, Abf, 128);

    // ---------------- layer 3 (in=128, H=2, C=128) ----------------
    gemm_bf16_wmma<<<4096, 256, 0, stream>>>(Abf, pWl3, XL, NN, 128, 256);
    gemm_bf16_wmma<<<4096, 256, 0, stream>>>(Abf, pWr3, XR, NN, 128, 256);
    hipMemsetAsync(maxk, 0, (size_t)NN * 16, stream);
    edge_logit_kernel<<<EB, 256, 0, stream>>>(XL, XR, eidx, edge_attr, ealoop, We3, att3, logit, 7);
    hipMemsetAsync(AGG, 0, (size_t)NN * 256 * 4, stream);
    seg_max_kernel<<<(EPn * 2) / 256, 256, 0, stream>>>(logit, eidx, maxk, 2);
    expsum_kernel<<<(EPn * 2) / 256, 256, 0, stream>>>(logit, eidx, maxk, denom, 2);
    aggregate_kernel<<<EB, 256, 0, stream>>>(XL, logit, denom, eidx, AGG, 2, 7);
    head_mean_bias_kernel<<<NC128, 256, 0, stream>>>(AGG, b3, XL, 128);
    bn_stats_kernel<<<128, 256, 0, stream>>>(XL, bnst, 128);
    bn_apply_kernel<<<NC128, 256, 0, stream>>>(XL, bnst, g3, be3, Abf, 128);  // GRU input

    // ---------------- GRU readout ----------------
    gru_kernel<<<BB / 16, 256, 0, stream>>>(Abf, pWih, pWhh, bih, bhh, outp);
}
